// CausalAttention_41961830482398
// MI455X (gfx1250) — compile-verified, matrix operand order checked by reading, not run on hardware
//
#include <hip/hip_runtime.h>

// ---------------------------------------------------------------------------
// MI455X (gfx1250): bf16 WMMA (16x16x32) everywhere, flash-attention with
// split-K across two key-halves (128 M-tiles x 2 halves = 256 blocks keeps
// the chip full while 64-row M-tiles cut K/V L2 streaming 4x vs round 2:
// every K/V fragment load feeds 4 WMMAs).
// Round 6: load_frag now issues two explicit uint4 (b128) loads per lane
// instead of eight dword loads — round 5 showed the vectorizer only forming
// ds_load_2addr_b32 pairs (+ address-math bloat) once the LICM-blocking
// opaque offset hid the alignment proof.
// ---------------------------------------------------------------------------

typedef __attribute__((ext_vector_type(16))) __bf16       v16bf;
typedef __attribute__((ext_vector_type(8)))  float        v8f;
typedef __attribute__((ext_vector_type(8)))  unsigned int v8u;

#define NTOK 8192
#define DIM  512
#define KC   32
#define RSQRT_D 0.04419417382415922f   // 1/sqrt(512)

__device__ __forceinline__ float nan2num(float x) {
  if (__builtin_isnan(x)) return 0.0f;
  if (__builtin_isinf(x)) return x > 0.0f ? 3.402823466e38f : -3.402823466e38f;
  return x;
}

__device__ __forceinline__ __bf16 to_bf16(float x) { return (__bf16)x; }

// Load one 16x32 bf16 fragment in the CDNA5 16-bit A/B layout:
//   lane&15 = row (M for A; N for B when the tile is stored [N][K]),
//   VGPR v holds K pair {k,k+1}, k = (v<4?0:16) + 8*(lane>=16) + 2*(v&3).
// Per lane that is exactly two 16-byte chunks: elements [kh,kh+8) and
// [16+kh,16+kh+8) -> two explicit b128 loads.
__device__ __forceinline__ v16bf load_frag(const __bf16* base, int ld) {
  const int lane = threadIdx.x & 31;
  const int r    = lane & 15;
  const int kh   = (lane >> 4) << 3;           // 0 or 8 (elements)
  const __bf16* p = base + (size_t)r * ld + kh;
  const uint4 lo = *(const uint4*)(p);
  const uint4 hi = *(const uint4*)(p + 16);
  v8u t;
  t[0] = lo.x; t[1] = lo.y; t[2] = lo.z; t[3] = lo.w;
  t[4] = hi.x; t[5] = hi.y; t[6] = hi.z; t[7] = hi.w;
  return __builtin_bit_cast(v16bf, t);
}

#define WMMA_BF16(A, B, C) \
  __builtin_amdgcn_wmma_f32_16x16x32_bf16(false, (A), false, (B), (short)0, (C), false, false)

// Cooperative copy of a 64x512 bf16 tile (64KB) global -> LDS, 256 threads.
__device__ __forceinline__ void stage_tile64(const __bf16* __restrict__ src,
                                             __bf16* __restrict__ dst) {
  const uint4* s = (const uint4*)src;
  uint4*       d = (uint4*)dst;
#pragma unroll
  for (int i = 0; i < 16; ++i)
    d[threadIdx.x + i * 256] = s[threadIdx.x + i * 256];
}

// ---------------------------------------------------------------------------
// Stage 1: nan_to_num + bf16 conversion of H and T, row norms of T, and
// zero-init of the distance row-sum accumulator.
// ---------------------------------------------------------------------------
__global__ void prep_rows_kernel(const float* __restrict__ H, const float* __restrict__ T,
                                 __bf16* __restrict__ Hb, __bf16* __restrict__ Tb,
                                 float* __restrict__ tt, float* __restrict__ rsum) {
  const int row = blockIdx.x;
  const int t   = threadIdx.x;
  const float* hp = H + (size_t)row * DIM;
  const float* tp = T + (size_t)row * DIM;
  __bf16* hb = Hb + (size_t)row * DIM;
  __bf16* tb = Tb + (size_t)row * DIM;
  float ss = 0.0f;
#pragma unroll
  for (int j = 0; j < 4; ++j) {
    const int c = j * 128 + t;
    const float hv = nan2num(hp[c]);
    hb[c] = to_bf16(hv);
    const float tv = nan2num(tp[c]);
    tb[c] = to_bf16(tv);
    ss += tv * tv;
  }
#pragma unroll
  for (int m = 16; m >= 1; m >>= 1) ss += __shfl_xor(ss, m, 32);
  __shared__ float red[4];
  if ((t & 31) == 0) red[t >> 5] = ss;
  __syncthreads();
  if (t == 0) {
    tt[row]   = red[0] + red[1] + red[2] + red[3];
    rsum[row] = 0.0f;
  }
}

__global__ void conv_bf16_kernel(const float* __restrict__ src, __bf16* __restrict__ dst, int n) {
  const int i = blockIdx.x * blockDim.x + threadIdx.x;
  if (i < n) dst[i] = to_bf16(src[i]);
}

// ---------------------------------------------------------------------------
// Stage 2: partial row sums of pairwise distances of T.
// Grid = (M/64, 2 key-halves); block = 8 waves owns a 64-row tile in LDS;
// each T-column fragment feeds 4 WMMAs. Results atomically added to rsum.
// ---------------------------------------------------------------------------
__global__ __launch_bounds__(256, 1) void
dist_partial_kernel(const __bf16* __restrict__ Tb, const float* __restrict__ tt,
                    float* __restrict__ rsum) {
  const int tid  = threadIdx.x;
  const int w    = tid >> 5;
  const int lane = tid & 31;
  const int hlf  = lane >> 4;
  const int l15  = lane & 15;
  const int m0   = blockIdx.x * 64;
  const int keyh = blockIdx.y;

  __shared__ __align__(16) __bf16 tLDS[64 * DIM];
  stage_tile64(Tb + (size_t)m0 * DIM, tLDS);
  __syncthreads();

  float ttm[4][8];
#pragma unroll
  for (int st = 0; st < 4; ++st)
#pragma unroll
    for (int v = 0; v < 8; ++v) ttm[st][v] = tt[m0 + st * 16 + v + 8 * hlf];

  float rs[4][8];
#pragma unroll
  for (int st = 0; st < 4; ++st)
#pragma unroll
    for (int v = 0; v < 8; ++v) rs[st][v] = 0.0f;

  unsigned toff = 0;   // asm-laundered per iteration: block LICM of A frags
  for (int it = keyh * 32; it < keyh * 32 + 32; ++it) {
    asm volatile("" : "+v"(toff));
    const __bf16* tbase =
        (const __bf16*)__builtin_assume_aligned((const char*)tLDS + toff, 16);
    const int kb = it * 128 + w * 16;
    if (kb + 128 < NTOK) __builtin_prefetch(Tb + (size_t)(kb + 128) * DIM, 0, 1);
    v8f acc[4];
    const v8f vz = {};
#pragma unroll
    for (int st = 0; st < 4; ++st) acc[st] = vz;
#pragma unroll
    for (int kc = 0; kc < 16; ++kc) {
      const v16bf b = load_frag(Tb + (size_t)kb * DIM + kc * KC, DIM);
#pragma unroll
      for (int st = 0; st < 4; ++st) {
        const v16bf a = load_frag(tbase + st * 16 * DIM + kc * KC, DIM);
        acc[st] = WMMA_BF16(a, b, acc[st]);
      }
    }
    const float ttn = tt[kb + l15];
#pragma unroll
    for (int st = 0; st < 4; ++st)
#pragma unroll
      for (int v = 0; v < 8; ++v) {
        const float d2 = ttm[st][v] + ttn - 2.0f * acc[st][v];
        rs[st][v] += sqrtf(fmaxf(d2, 0.0f) + 1e-8f);
      }
  }
  // reduce over the 16 lanes (N) of each half-wave, then global accumulate
#pragma unroll
  for (int m = 8; m >= 1; m >>= 1)
#pragma unroll
    for (int st = 0; st < 4; ++st)
#pragma unroll
      for (int v = 0; v < 8; ++v) rs[st][v] += __shfl_xor(rs[st][v], m, 32);
  if (l15 == 0) {
#pragma unroll
    for (int st = 0; st < 4; ++st)
#pragma unroll
      for (int v = 0; v < 8; ++v)
        atomicAdd(&rsum[m0 + st * 16 + v + 8 * hlf], rs[st][v]);
  }
}

// ---------------------------------------------------------------------------
// GEMM: out = A(bf16,[M][512]) @ W(bf16,[N][512])^T + bias.
// MODE 0: bf16 row-major out; MODE 1: bf16 transposed out [N][M] (for V);
// MODE 2: f32 row-major out (final projection).
// ---------------------------------------------------------------------------
template <int MODE>
__global__ __launch_bounds__(256, 1) void
gemm_bias_kernel(const __bf16* __restrict__ A, const __bf16* __restrict__ W,
                 const float* __restrict__ bias, void* __restrict__ out,
                 int M, int Nout) {
  const int tid  = threadIdx.x;
  const int w    = tid >> 5;
  const int lane = tid & 31;
  const int hlf  = lane >> 4;
  const int l15  = lane & 15;
  const int m0   = blockIdx.x * 16;
  const int n0   = (blockIdx.y * 8 + w) * 16;

  __shared__ __align__(16) __bf16 aLDS[16 * DIM];
  {
    const uint4* s = (const uint4*)(A + (size_t)m0 * DIM);
    uint4*       d = (uint4*)aLDS;
#pragma unroll
    for (int i = 0; i < 4; ++i)
      d[tid + i * 256] = s[tid + i * 256];
  }
  __syncthreads();

  v8f acc = {};
#pragma unroll
  for (int kc = 0; kc < 16; ++kc) {
    const v16bf a = load_frag(aLDS + kc * KC, DIM);
    const v16bf b = load_frag(W + (size_t)n0 * DIM + kc * KC, DIM);
    acc = WMMA_BF16(a, b, acc);
  }
  const float bn = bias[n0 + l15];
#pragma unroll
  for (int v = 0; v < 8; ++v) {
    const float  val = acc[v] + bn;
    const size_t m   = m0 + v + 8 * hlf;
    const size_t n   = n0 + l15;
    if (MODE == 0)      ((__bf16*)out)[m * Nout + n] = to_bf16(val);
    else if (MODE == 1) ((__bf16*)out)[n * (size_t)M + m] = to_bf16(val);
    else                ((float*)out)[m * Nout + n] = val;
  }
}

// ---------------------------------------------------------------------------
// Flash attention, split-K: grid = (M/64, 2 key-halves). Block = 8 waves owns
// a 64-row query tile (LDS). Per 128-key step: wave w computes S for its
// 16-key tile against all 4 row sub-tiles (K fragment -> 4 WMMAs), online
// softmax (4 threads/row over 64 rows), then P@V on the wave's 64-dim slice
// (V fragment -> 4 WMMAs). Unnormalized O (f32) + per-row (m,l) written out.
// ---------------------------------------------------------------------------
__global__ __launch_bounds__(256, 1) void
flash_attn_kernel(const __bf16* __restrict__ Qb, const __bf16* __restrict__ Kb,
                  const __bf16* __restrict__ Vt, float* __restrict__ Opart,
                  float* __restrict__ mpart, float* __restrict__ lpart) {
  const int tid  = threadIdx.x;
  const int w    = tid >> 5;
  const int lane = tid & 31;
  const int hlf  = lane >> 4;
  const int l15  = lane & 15;
  const int m0   = blockIdx.x * 64;
  const int keyh = blockIdx.y;
  const int d0   = w * 64;

  __shared__ __align__(16) __bf16 qLDS[64 * DIM];     // 64KB
  __shared__ __align__(16) float  sLDS[64 * 128];     // 32KB
  __shared__ __align__(16) __bf16 pLDS[64 * 128];     // 16KB
  __shared__ float mLDS[64], lLDS[64], aLDS[64];

  if (tid < 64) { mLDS[tid] = -3.0e38f; lLDS[tid] = 0.0f; }
  stage_tile64(Qb + (size_t)m0 * DIM, qLDS);

  const v8f vz = {};
  v8f o[4][4];
#pragma unroll
  for (int st = 0; st < 4; ++st)
#pragma unroll
    for (int t = 0; t < 4; ++t) o[st][t] = vz;
  __syncthreads();

  unsigned qoff = 0;   // asm-laundered per iteration: block LICM of A frags
  for (int it = keyh * 32; it < keyh * 32 + 32; ++it) {
    asm volatile("" : "+v"(qoff));
    const __bf16* qbase =
        (const __bf16*)__builtin_assume_aligned((const char*)qLDS + qoff, 16);
    // ---- S = (Q @ K^T)/sqrt(d): wave's 16-key tile x 4 row sub-tiles ----
    const int kb = it * 128 + w * 16;
    if (kb + 128 < NTOK) __builtin_prefetch(Kb + (size_t)(kb + 128) * DIM, 0, 1);
    v8f s[4];
#pragma unroll
    for (int st = 0; st < 4; ++st) s[st] = vz;
#pragma unroll
    for (int kc = 0; kc < 16; ++kc) {
      const v16bf b = load_frag(Kb + (size_t)kb * DIM + kc * KC, DIM);
#pragma unroll
      for (int st = 0; st < 4; ++st) {
        const v16bf a = load_frag(qbase + st * 16 * DIM + kc * KC, DIM);
        s[st] = WMMA_BF16(a, b, s[st]);
      }
    }
#pragma unroll
    for (int st = 0; st < 4; ++st)
#pragma unroll
      for (int v = 0; v < 8; ++v)
        sLDS[(st * 16 + v + 8 * hlf) * 128 + w * 16 + l15] = s[st][v] * RSQRT_D;
    __syncthreads();

    // ---- online softmax over the 128-key block: 4 threads/row, 64 rows ----
    {
      const int r = tid >> 2, sub = tid & 3;
      const float* srow = sLDS + r * 128 + sub * 32;
      float lm = -3.0e38f;
#pragma unroll
      for (int j = 0; j < 32; ++j) lm = fmaxf(lm, srow[j]);
#pragma unroll
      for (int m = 2; m >= 1; m >>= 1) lm = fmaxf(lm, __shfl_xor(lm, m, 32));
      const float m_prev = mLDS[r];
      const float m_new  = fmaxf(m_prev, lm);
      const float alpha  = __expf(m_prev - m_new);
      float ps = 0.0f;
      __bf16* prow = pLDS + r * 128 + sub * 32;
#pragma unroll
      for (int j = 0; j < 32; ++j) {
        const float e = __expf(srow[j] - m_new);
        prow[j] = to_bf16(e);
        ps += e;
      }
#pragma unroll
      for (int m = 2; m >= 1; m >>= 1) ps += __shfl_xor(ps, m, 32);
      if (sub == 0) { lLDS[r] = lLDS[r] * alpha + ps; mLDS[r] = m_new; aLDS[r] = alpha; }
    }
    __syncthreads();

    // ---- rescale O, then O += P @ V (all sub-tiles) on 64-dim slice ----
#pragma unroll
    for (int st = 0; st < 4; ++st) {
      float a8[8];
#pragma unroll
      for (int v = 0; v < 8; ++v) a8[v] = aLDS[st * 16 + v + 8 * hlf];
#pragma unroll
      for (int t = 0; t < 4; ++t)
#pragma unroll
        for (int v = 0; v < 8; ++v) o[st][t][v] *= a8[v];
    }
#pragma unroll
    for (int kc = 0; kc < 4; ++kc) {
      v16bf pa[4];
#pragma unroll
      for (int st = 0; st < 4; ++st)
        pa[st] = load_frag(pLDS + st * 16 * 128 + kc * KC, 128);
#pragma unroll
      for (int t = 0; t < 4; ++t) {
        const v16bf vb = load_frag(Vt + (size_t)(d0 + t * 16) * NTOK +
                                   (size_t)it * 128 + kc * KC, NTOK);
#pragma unroll
        for (int st = 0; st < 4; ++st) o[st][t] = WMMA_BF16(pa[st], vb, o[st][t]);
      }
    }
    __syncthreads();
  }

  // ---- write unnormalized partial O (f32) + row stats ----
  float* Obase = Opart + (size_t)keyh * NTOK * DIM;
#pragma unroll
  for (int st = 0; st < 4; ++st)
#pragma unroll
    for (int v = 0; v < 8; ++v) {
      const size_t row = (size_t)(m0 + st * 16 + v + 8 * hlf) * DIM;
#pragma unroll
      for (int t = 0; t < 4; ++t)
        Obase[row + d0 + t * 16 + l15] = o[st][t][v];
    }
  if (tid < 64) {
    mpart[(size_t)keyh * NTOK + m0 + tid] = mLDS[tid];
    lpart[(size_t)keyh * NTOK + m0 + tid] = lLDS[tid];
  }
}

// ---------------------------------------------------------------------------
// Combine the two key-halves, apply softmax normalization and the distance
// row-scale, emit bf16 AV for the output projection. One block per row.
// ---------------------------------------------------------------------------
__global__ void combine_kernel(const float* __restrict__ Opart,
                               const float* __restrict__ mpart,
                               const float* __restrict__ lpart,
                               const float* __restrict__ rsum,
                               __bf16* __restrict__ AVb) {
  const int row = blockIdx.x;
  const float m0v = mpart[row], m1v = mpart[NTOK + row];
  const float l0v = lpart[row], l1v = lpart[NTOK + row];
  const float M   = fmaxf(m0v, m1v);
  const float a0  = __expf(m0v - M);
  const float a1  = __expf(m1v - M);
  const float den = a0 * l0v + a1 * l1v;
  const float sc  = 1.0f / (1.0f + rsum[row] * (1.0f / (float)NTOK));
  const float f   = sc / den;
  const float* O0 = Opart + (size_t)row * DIM;
  const float* O1 = Opart + (size_t)NTOK * DIM + (size_t)row * DIM;
  __bf16* dst = AVb + (size_t)row * DIM;
#pragma unroll
  for (int j = 0; j < 2; ++j) {
    const int c = threadIdx.x + j * 256;
    dst[c] = to_bf16(f * (a0 * O0[c] + a1 * O1[c]));
  }
}

// ---------------------------------------------------------------------------
extern "C" void kernel_launch(void* const* d_in, const int* in_sizes, int n_in,
                              void* d_out, int out_size, void* d_ws, size_t ws_size,
                              hipStream_t stream) {
  const float* H  = (const float*)d_in[0];
  const float* T  = (const float*)d_in[1];
  const float* Wq = (const float*)d_in[2];
  const float* bq = (const float*)d_in[3];
  const float* Wk = (const float*)d_in[4];
  const float* bk = (const float*)d_in[5];
  const float* Wv = (const float*)d_in[6];
  const float* bv = (const float*)d_in[7];
  const float* Wo = (const float*)d_in[8];
  const float* bo = (const float*)d_in[9];
  float* out = (float*)d_out;

  char*  ws  = (char*)d_ws;
  size_t off = 0;
  auto take  = [&](size_t bytes) {
    char* p = ws + off;
    off = (off + bytes + 255) & ~(size_t)255;
    return p;
  };
  const size_t nd2 = (size_t)NTOK * DIM * 2;   // bf16 [N][DIM]
  const size_t nd4 = (size_t)NTOK * DIM * 4;   // f32  [N][DIM]
  const size_t ww2 = (size_t)DIM * DIM * 2;    // bf16 [DIM][DIM]
  __bf16* Hb  = (__bf16*)take(nd2);
  __bf16* Tb  = (__bf16*)take(nd2);
  __bf16* Qb  = (__bf16*)take(nd2);
  __bf16* Kb  = (__bf16*)take(nd2);
  __bf16* Vt  = (__bf16*)take(nd2);            // stored transposed [DIM][N]
  __bf16* AVb = (__bf16*)take(nd2);
  __bf16* Wqb = (__bf16*)take(ww2);
  __bf16* Wkb = (__bf16*)take(ww2);
  __bf16* Wvb = (__bf16*)take(ww2);
  __bf16* Wob = (__bf16*)take(ww2);
  float*  Opart = (float*)take(2 * nd4);       // two key-half partials
  float*  mpart = (float*)take(2 * (size_t)NTOK * 4);
  float*  lpart = (float*)take(2 * (size_t)NTOK * 4);
  float*  tt    = (float*)take((size_t)NTOK * 4);
  float*  rsum  = (float*)take((size_t)NTOK * 4);

  // Stage 1: convert + norms (+ rsum zero-init)
  prep_rows_kernel<<<NTOK, 128, 0, stream>>>(H, T, Hb, Tb, tt, rsum);
  const int wn = DIM * DIM;
  conv_bf16_kernel<<<(wn + 255) / 256, 256, 0, stream>>>(Wq, Wqb, wn);
  conv_bf16_kernel<<<(wn + 255) / 256, 256, 0, stream>>>(Wk, Wkb, wn);
  conv_bf16_kernel<<<(wn + 255) / 256, 256, 0, stream>>>(Wv, Wvb, wn);
  conv_bf16_kernel<<<(wn + 255) / 256, 256, 0, stream>>>(Wo, Wob, wn);

  // Stage 2: distance row sums (partial, atomically accumulated)
  dist_partial_kernel<<<dim3(NTOK / 64, 2), 256, 0, stream>>>(Tb, tt, rsum);

  // Stage 3: Q/K/V projections (V transposed for PV B-fragment layout)
  const dim3 ggrid(NTOK / 16, DIM / 128);
  gemm_bias_kernel<0><<<ggrid, 256, 0, stream>>>(Hb, Wqb, bq, Qb, NTOK, DIM);
  gemm_bias_kernel<0><<<ggrid, 256, 0, stream>>>(Hb, Wkb, bk, Kb, NTOK, DIM);
  gemm_bias_kernel<1><<<ggrid, 256, 0, stream>>>(Hb, Wvb, bv, Vt, NTOK, DIM);

  // Stage 4: fused attention, split over 2 key-halves
  flash_attn_kernel<<<dim3(NTOK / 64, 2), 256, 0, stream>>>(Qb, Kb, Vt,
                                                            Opart, mpart, lpart);

  // Stage 5: merge halves + softmax normalization + distance row scale
  combine_kernel<<<NTOK, 256, 0, stream>>>(Opart, mpart, lpart, rsum, AVb);

  // Stage 6: output projection, f32 result
  gemm_bias_kernel<2><<<ggrid, 256, 0, stream>>>(AVb, Wob, bo, out, NTOK, DIM);
}
